// Attention_1443109011815
// MI455X (gfx1250) — compile-verified, matrix operand order checked by reading, not run on hardware
//
#include <hip/hip_runtime.h>
#include <hip/hip_bf16.h>

// ---------------------------------------------------------------------------
// Problem constants (match reference)
// ---------------------------------------------------------------------------
#define BATCH 16
#define NTOK  1024
#define INP   256
#define HEADS 8
#define DHEAD 32
#define INNER 256
#define OUP   256

typedef __attribute__((ext_vector_type(16))) __bf16 v16bf;
typedef __attribute__((ext_vector_type(8)))  __bf16 v8bf;
typedef __attribute__((ext_vector_type(8)))  float  v8f;

// ---------------------------------------------------------------------------
// WMMA helpers (gfx1250 / CDNA5, wave32)
//
// C/D layout (ISA 7.12.2, 32-bit 16x16): VGPR r, lane L -> row = r + 8*(L>>4),
// col = L & 15.
// A layout (16-bit 16x32): lane L holds row (L&15); within the 16-element
// fragment, elems 0..7 are K = 8*(L>>4)+j, elems 8..15 are K = 16+8*(L>>4)+j.
// We store the "B" operand sources row-major as [N x K] panels so the same
// loader serves both operands.
// ---------------------------------------------------------------------------
__device__ __forceinline__ v16bf load_frag(const __bf16* __restrict__ base, int ld) {
  const int lane = threadIdx.x & 31;
  const int r    = lane & 15;
  const int hi8  = (lane >> 4) << 3;           // 0 or 8
  const __bf16* p = base + (size_t)r * ld;
  v8bf lo = *(const v8bf*)(p + hi8);
  v8bf hh = *(const v8bf*)(p + 16 + hi8);
  v16bf f;
#pragma unroll
  for (int i = 0; i < 8; ++i) { f[i] = lo[i]; f[i + 8] = hh[i]; }
  return f;
}

__device__ __forceinline__ v8f wmma_bf16(v16bf a, v16bf b, v8f c) {
  return __builtin_amdgcn_wmma_f32_16x16x32_bf16(
      /*neg_a=*/false, a, /*neg_b=*/false, b,
      /*c_mod=*/(short)0, c, /*reuse_a=*/false, /*reuse_b=*/false);
}

// ---------------------------------------------------------------------------
// Conversion / layout kernels
// ---------------------------------------------------------------------------
__global__ void cvt_f32_bf16(const float* __restrict__ in, __bf16* __restrict__ out, int n) {
  int i = blockIdx.x * blockDim.x + threadIdx.x;
  if (i < n) out[i] = (__bf16)in[i];
}

// in[R][C] (f32, row-major) -> out[C][R] (bf16)
__global__ void transpose_to_bf16(const float* __restrict__ in, __bf16* __restrict__ out,
                                  int R, int C) {
  int i = blockIdx.x * blockDim.x + threadIdx.x;
  if (i < R * C) {
    int r = i / C, c = i - r * C;
    out[(size_t)c * R + r] = (__bf16)in[i];
  }
}

// bias_all[h][i][j] = bias_table[rel_index[i*N+j]][h]   (bf16, dense)
__global__ void bias_gather(const float* __restrict__ table, const int* __restrict__ ridx,
                            __bf16* __restrict__ bias_all) {
  int p = blockIdx.x * blockDim.x + threadIdx.x;
  if (p < NTOK * NTOK) {
    int idx = ridx[p];
#pragma unroll
    for (int h = 0; h < HEADS; ++h)
      bias_all[(size_t)h * NTOK * NTOK + p] = (__bf16)table[idx * HEADS + h];
  }
}

// ---------------------------------------------------------------------------
// QKV projection GEMM: [16384 x 256] @ [256 x 768] -> q/k/vT (bf16)
// 256 threads = 8 waves; each wave computes a 32x64 output tile:
// 2 A-fragments x 4 B-fragments -> 8 WMMAs per k-step, B reused 2x.
// ---------------------------------------------------------------------------
__global__ __launch_bounds__(256) void qkv_gemm(const __bf16* __restrict__ xb,
                                                const __bf16* __restrict__ wT,  // [768][256]
                                                __bf16* __restrict__ qd,        // [b,h,n,32]
                                                __bf16* __restrict__ kd,        // [b,h,n,32]
                                                __bf16* __restrict__ vTd) {     // [b,h,32,n]
  const int wave = threadIdx.x >> 5;
  const int lane = threadIdx.x & 31;
  const int hi   = lane >> 4;
  const int cl   = lane & 15;
  const int row0 = blockIdx.x * 256 + wave * 32;
  const int col0 = blockIdx.y * 64;

  v8f acc[2][4];
#pragma unroll
  for (int s = 0; s < 2; ++s)
#pragma unroll
    for (int t = 0; t < 4; ++t) acc[s][t] = (v8f){};

#pragma unroll
  for (int k0 = 0; k0 < INP; k0 += 32) {
    v16bf a0 = load_frag(xb + (size_t)row0 * INP + k0, INP);
    v16bf a1 = load_frag(xb + (size_t)(row0 + 16) * INP + k0, INP);
#pragma unroll
    for (int t = 0; t < 4; ++t) {
      v16bf bf = load_frag(wT + (size_t)(col0 + 16 * t) * INP + k0, INP);
      acc[0][t] = wmma_bf16(a0, bf, acc[0][t]);
      acc[1][t] = wmma_bf16(a1, bf, acc[1][t]);
    }
  }

#pragma unroll
  for (int s = 0; s < 2; ++s) {
#pragma unroll
    for (int t = 0; t < 4; ++t) {
#pragma unroll
      for (int r = 0; r < 8; ++r) {
        int rg = row0 + 16 * s + r + 8 * hi;  // global row in [0, B*N)
        int cg = col0 + 16 * t + cl;          // global col in [0, 768)
        int b = rg >> 10, n = rg & (NTOK - 1);
        int which = cg >> 8, rem = cg & 255;
        int h = rem >> 5, d = rem & 31;
        __bf16 val = (__bf16)acc[s][t][r];
        size_t bh = (size_t)(b * HEADS + h);
        if (which == 0)      qd[(bh * NTOK + n) * DHEAD + d] = val;
        else if (which == 1) kd[(bh * NTOK + n) * DHEAD + d] = val;
        else                 vTd[(bh * DHEAD + d) * NTOK + n] = val;
      }
    }
  }
}

// ---------------------------------------------------------------------------
// Flash attention: grid (B*H, N/64), 128 threads = 4 waves, 16 rows/wave.
// ---------------------------------------------------------------------------
__global__ __launch_bounds__(128) void attn_kernel(const __bf16* __restrict__ q,
                                                   const __bf16* __restrict__ k,
                                                   const __bf16* __restrict__ vT,
                                                   const __bf16* __restrict__ bias_all,
                                                   __bf16* __restrict__ attn_out) {
  __shared__ __align__(16) __bf16 pbuf[4 * 16 * 64];   // per-wave 16x64 P staging

  const int wave = threadIdx.x >> 5;
  const int lane = threadIdx.x & 31;
  const int hi   = lane >> 4;
  const int cl   = lane & 15;

  const int bh = blockIdx.x;                 // b*HEADS + h
  const int b  = bh >> 3;
  const int h  = bh & 7;
  const int r0 = blockIdx.y * 64 + wave * 16;

  const __bf16* qbh   = q  + (size_t)bh * NTOK * DHEAD;
  const __bf16* kbh   = k  + (size_t)bh * NTOK * DHEAD;
  const __bf16* vbh   = vT + (size_t)bh * DHEAD * NTOK;
  const __bf16* biash = bias_all + (size_t)h * NTOK * NTOK;

  // Q fragment for this wave's 16 rows (K = DHEAD = 32, one fragment)
  v16bf qf = load_frag(qbh + (size_t)r0 * DHEAD, DHEAD);

  v8f O0 = (v8f){}, O1 = (v8f){};
  float m[8], l[8];
#pragma unroll
  for (int r = 0; r < 8; ++r) { m[r] = -1e30f; l[r] = 0.0f; }
  const float scale = 0.17677669529663687f;  // 1/sqrt(32)

  __bf16* pw = pbuf + wave * 16 * 64;

  for (int cb = 0; cb < NTOK; cb += 64) {
    // Prefetch next column block while this block's softmax runs
    // (__builtin_prefetch lowers to global_prefetch_b8 on gfx1250).
    if (cb + 64 < NTOK) {
      __builtin_prefetch(kbh + (size_t)(cb + 64) * DHEAD + lane * 64, 0, 1);
      __builtin_prefetch(vbh + (size_t)lane * NTOK + cb + 64, 0, 1);
      __builtin_prefetch(biash + (size_t)(r0 + cl) * NTOK + cb + 64, 0, 1);
    }

    // S = q @ k^T for 16x64 block (4 WMMAs)
    v8f S[4];
#pragma unroll
    for (int t = 0; t < 4; ++t) {
      v16bf kf = load_frag(kbh + (size_t)(cb + 16 * t) * DHEAD, DHEAD);
      v8f z = (v8f){};
      S[t] = wmma_bf16(qf, kf, z);
    }
    // scale + relative-position bias (C-layout addressing)
#pragma unroll
    for (int t = 0; t < 4; ++t) {
      int cc = cb + 16 * t + cl;
#pragma unroll
      for (int r = 0; r < 8; ++r) {
        int rr = r0 + r + 8 * hi;
        float bv = (float)biash[(size_t)rr * NTOK + cc];
        S[t][r] = S[t][r] * scale + bv;
      }
    }
    // online softmax update (rows live in 16-lane halves)
#pragma unroll
    for (int r = 0; r < 8; ++r) {
      float bm = fmaxf(fmaxf(S[0][r], S[1][r]), fmaxf(S[2][r], S[3][r]));
      bm = fmaxf(bm, __shfl_xor(bm, 1, 32));
      bm = fmaxf(bm, __shfl_xor(bm, 2, 32));
      bm = fmaxf(bm, __shfl_xor(bm, 4, 32));
      bm = fmaxf(bm, __shfl_xor(bm, 8, 32));
      float mn = fmaxf(m[r], bm);
      float alpha = __expf(m[r] - mn);
      float rs = 0.0f;
#pragma unroll
      for (int t = 0; t < 4; ++t) {
        float p = __expf(S[t][r] - mn);
        S[t][r] = p;
        rs += p;
      }
      rs += __shfl_xor(rs, 1, 32);
      rs += __shfl_xor(rs, 2, 32);
      rs += __shfl_xor(rs, 4, 32);
      rs += __shfl_xor(rs, 8, 32);
      l[r] = l[r] * alpha + rs;
      m[r] = mn;
      O0[r] *= alpha;
      O1[r] *= alpha;
    }
    // stage P (C-layout) -> LDS row-major 16x64 (bf16), re-fragment as A
#pragma unroll
    for (int t = 0; t < 4; ++t)
#pragma unroll
      for (int r = 0; r < 8; ++r)
        pw[(r + 8 * hi) * 64 + 16 * t + cl] = (__bf16)S[t][r];
    __syncthreads();
    // O += P @ V : K = 64 (2 steps), N = 32 (2 tiles)
#pragma unroll
    for (int ks = 0; ks < 2; ++ks) {
      v16bf pf = load_frag(pw + 32 * ks, 64);
      v16bf vf0 = load_frag(vbh + (size_t)0  * NTOK + cb + 32 * ks, NTOK);
      v16bf vf1 = load_frag(vbh + (size_t)16 * NTOK + cb + 32 * ks, NTOK);
      O0 = wmma_bf16(pf, vf0, O0);
      O1 = wmma_bf16(pf, vf1, O1);
    }
    __syncthreads();
  }

  // normalize + store to attn_out[b, n, h*32 + d] (bf16, K-contiguous for out proj)
#pragma unroll
  for (int r = 0; r < 8; ++r) {
    int rr = r0 + r + 8 * hi;
    float inv = 1.0f / l[r];
    size_t rowbase = ((size_t)b * NTOK + rr) * INNER + h * DHEAD;
    attn_out[rowbase + cl]      = (__bf16)(O0[r] * inv);
    attn_out[rowbase + 16 + cl] = (__bf16)(O1[r] * inv);
  }
}

// ---------------------------------------------------------------------------
// Output projection: [16384 x 256] @ [256 x 256] + b_out -> f32
// Same 32x64 per-wave tiling as qkv_gemm.
// ---------------------------------------------------------------------------
__global__ __launch_bounds__(256) void out_gemm(const __bf16* __restrict__ ab,
                                                const __bf16* __restrict__ wT,   // [256][256]
                                                const float* __restrict__ bout,
                                                float* __restrict__ out) {
  const int wave = threadIdx.x >> 5;
  const int lane = threadIdx.x & 31;
  const int hi   = lane >> 4;
  const int cl   = lane & 15;
  const int row0 = blockIdx.x * 256 + wave * 32;
  const int col0 = blockIdx.y * 64;

  v8f acc[2][4];
#pragma unroll
  for (int s = 0; s < 2; ++s)
#pragma unroll
    for (int t = 0; t < 4; ++t) acc[s][t] = (v8f){};

#pragma unroll
  for (int k0 = 0; k0 < INNER; k0 += 32) {
    v16bf a0 = load_frag(ab + (size_t)row0 * INNER + k0, INNER);
    v16bf a1 = load_frag(ab + (size_t)(row0 + 16) * INNER + k0, INNER);
#pragma unroll
    for (int t = 0; t < 4; ++t) {
      v16bf bf = load_frag(wT + (size_t)(col0 + 16 * t) * INNER + k0, INNER);
      acc[0][t] = wmma_bf16(a0, bf, acc[0][t]);
      acc[1][t] = wmma_bf16(a1, bf, acc[1][t]);
    }
  }

#pragma unroll
  for (int s = 0; s < 2; ++s) {
#pragma unroll
    for (int t = 0; t < 4; ++t) {
#pragma unroll
      for (int r = 0; r < 8; ++r) {
        int rg = row0 + 16 * s + r + 8 * hi;
        int cg = col0 + 16 * t + cl;
        out[(size_t)rg * OUP + cg] = acc[s][t][r] + bout[cg];
      }
    }
  }
}

// ---------------------------------------------------------------------------
// Workspace layout (bytes)
// ---------------------------------------------------------------------------
static constexpr size_t OFF_XB    = 0;                                  // 8 MB
static constexpr size_t OFF_WQKVT = OFF_XB    + (size_t)BATCH * NTOK * INP * 2;
static constexpr size_t OFF_WOUTT = OFF_WQKVT + (size_t)(3 * INNER) * INP * 2;
static constexpr size_t OFF_Q     = OFF_WOUTT + (size_t)OUP * INNER * 2;
static constexpr size_t OFF_K     = OFF_Q     + (size_t)BATCH * HEADS * NTOK * DHEAD * 2;
static constexpr size_t OFF_VT    = OFF_K     + (size_t)BATCH * HEADS * NTOK * DHEAD * 2;
static constexpr size_t OFF_AO    = OFF_VT    + (size_t)BATCH * HEADS * NTOK * DHEAD * 2;
static constexpr size_t OFF_BIAS  = OFF_AO    + (size_t)BATCH * NTOK * INNER * 2;

extern "C" void kernel_launch(void* const* d_in, const int* in_sizes, int n_in,
                              void* d_out, int out_size, void* d_ws, size_t ws_size,
                              hipStream_t stream) {
  const float* x          = (const float*)d_in[0];
  const float* w_qkv      = (const float*)d_in[1];
  const float* w_out      = (const float*)d_in[2];
  const float* b_out      = (const float*)d_in[3];
  const float* bias_table = (const float*)d_in[4];
  const int*   rel_index  = (const int*)d_in[5];

  char* ws = (char*)d_ws;
  __bf16* xb     = (__bf16*)(ws + OFF_XB);
  __bf16* wqkvT  = (__bf16*)(ws + OFF_WQKVT);
  __bf16* woutT  = (__bf16*)(ws + OFF_WOUTT);
  __bf16* qd     = (__bf16*)(ws + OFF_Q);
  __bf16* kd     = (__bf16*)(ws + OFF_K);
  __bf16* vTd    = (__bf16*)(ws + OFF_VT);
  __bf16* ao     = (__bf16*)(ws + OFF_AO);
  __bf16* biasA  = (__bf16*)(ws + OFF_BIAS);
  float*  out    = (float*)d_out;

  // 1) precision conversion + weight transposes
  {
    int n = BATCH * NTOK * INP;
    cvt_f32_bf16<<<(n + 255) / 256, 256, 0, stream>>>(x, xb, n);
  }
  transpose_to_bf16<<<(INP * 3 * INNER + 255) / 256, 256, 0, stream>>>(w_qkv, wqkvT, INP, 3 * INNER);
  transpose_to_bf16<<<(INNER * OUP + 255) / 256, 256, 0, stream>>>(w_out, woutT, INNER, OUP);

  // 2) dense relative-position bias table [H, N, N]
  bias_gather<<<(NTOK * NTOK + 255) / 256, 256, 0, stream>>>(bias_table, rel_index, biasA);

  // 3) QKV projection (WMMA)
  {
    dim3 grid((BATCH * NTOK) / 256, (3 * INNER) / 64);
    qkv_gemm<<<grid, 256, 0, stream>>>(xb, wqkvT, qd, kd, vTd);
  }

  // 4) flash attention (WMMA)
  {
    dim3 grid(BATCH * HEADS, NTOK / 64);
    attn_kernel<<<grid, 128, 0, stream>>>(qd, kd, vTd, biasA, ao);
  }

  // 5) output projection + bias (WMMA)
  {
    dim3 grid((BATCH * NTOK) / 256, OUP / 64);
    out_gemm<<<grid, 256, 0, stream>>>(ao, woutT, b_out, out);
  }
}
